// SameDenominatorLinkPredictor_15522011808348
// MI455X (gfx1250) — compile-verified, matrix operand order checked by reading, not run on hardware
//
#include <hip/hip_runtime.h>
#include <hip/hip_bf16.h>

// ---------------------------------------------------------------------------
// GCN link predictor for gfx1250 (MI455X).
//   encoder: 2x GCNConv (deg^-1/2 sym norm, self loops) + fc -> z[N,64]
//   decoder: gather pairs -> MLP(128->128->64->1) -> sigmoid
// All K=128 GEMMs use v_wmma_f32_16x16x32_f16. One wave owns a 16-row block,
// keeps A k-fragments in VGPRs, iterates over N-tiles (A/gather traffic /8).
// ---------------------------------------------------------------------------

typedef __attribute__((ext_vector_type(16))) _Float16 v16h;
typedef __attribute__((ext_vector_type(8)))  float    v8f;

#define NN 100000     // nodes
#define NE 1600000    // edges
#define NQ 500000     // query pairs
#define F_IN 7
#define HID 128
#define EMB 64

union Frag16 { v16h h; float4 f4[2]; };

// 16-bit A-matrix 16x32 fragment layout (ISA 7.12.2):
// lanes 0-15 : row M=lane,    halves = K[k0+0..7],  K[k0+16..23]
// lanes 16-31: row M=lane-16, halves = K[k0+8..15], K[k0+24..31]
__device__ __forceinline__ v16h load_frag(const _Float16* __restrict__ rowPtr,
                                          int k0, int lane) {
  const int off = (lane >> 4) << 3;  // 0 or 8
  Frag16 u;
  u.f4[0] = *reinterpret_cast<const float4*>(rowPtr + k0 + off);
  u.f4[1] = *reinterpret_cast<const float4*>(rowPtr + k0 + off + 16);
  return u.h;
}

// ---------------------------------------------------------------------------
// small elementwise / scatter kernels
// ---------------------------------------------------------------------------
__global__ void k_fill1(float* p, int n) {
  int i = blockIdx.x * blockDim.x + threadIdx.x;
  if (i < n) p[i] = 1.0f;
}

__global__ void k_deg_accum(const int* __restrict__ ei, float* __restrict__ deg) {
  int e = blockIdx.x * blockDim.x + threadIdx.x;
  if (e < NE) atomicAdd(&deg[ei[NE + e]], 1.0f);
}

__global__ void k_rsqrt(float* p, int n) {
  int i = blockIdx.x * blockDim.x + threadIdx.x;
  if (i < n) p[i] = rsqrtf(p[i]);
}

// Wt[n*K + k] = (f16) W[k*N + n]   (pre-transpose weights -> contiguous B frags)
__global__ void k_cvtT(const float* __restrict__ W, _Float16* __restrict__ Wt,
                       int K, int N) {
  int i = blockIdx.x * blockDim.x + threadIdx.x;
  if (i < K * N) {
    int n = i / K, k = i % K;
    Wt[i] = (_Float16)W[k * N + n];
  }
}

// h[n][j] = sum_{k<7} x[n][k] * W1[k][j]   (tiny K, plain FMA)
__global__ void k_in_gemm(const float* __restrict__ x, const float* __restrict__ W1,
                          float* __restrict__ h) {
  int t = blockIdx.x * blockDim.x + threadIdx.x;
  int n = t >> 7, j = t & 127;
  if (n < NN) {
    float a = 0.0f;
#pragma unroll
    for (int k = 0; k < F_IN; ++k) a = fmaf(x[n * F_IN + k], W1[k * HID + j], a);
    h[t] = a;
  }
}

// agg[n][j] = h[n][j]*dinv[n]^2 + b[j]   (self-loop + bias init)
__global__ void k_self_init(const float* __restrict__ h, const float* __restrict__ dinv,
                            const float* __restrict__ b, float* __restrict__ agg) {
  int t = blockIdx.x * blockDim.x + threadIdx.x;
  int n = t >> 7, j = t & 127;
  if (n < NN) {
    float di = dinv[n];
    agg[t] = h[t] * di * di + b[j];
  }
}

// per (edge, 4-feature group): agg[dst] += h[src]*dinv[src]*dinv[dst]
__global__ void k_scatter(const float* __restrict__ h, const float* __restrict__ dinv,
                          const int* __restrict__ ei, float* __restrict__ agg) {
  long t = (long)blockIdx.x * blockDim.x + threadIdx.x;
  long e = t >> 5;
  int seg = ((int)t & 31) << 2;
  if (e < NE) {
    int s = ei[e], d = ei[NE + e];
    float norm = dinv[s] * dinv[d];
    float4 m = *reinterpret_cast<const float4*>(h + (long)s * HID + seg);
    float* o = agg + (long)d * HID + seg;
    atomicAdd(o + 0, m.x * norm);
    atomicAdd(o + 1, m.y * norm);
    atomicAdd(o + 2, m.z * norm);
    atomicAdd(o + 3, m.w * norm);
  }
}

__global__ void k_relu_cvt(const float* __restrict__ agg, _Float16* __restrict__ hh,
                           long n) {
  long i = (long)blockIdx.x * blockDim.x + threadIdx.x;
  if (i < n) hh[i] = (_Float16)fmaxf(agg[i], 0.0f);
}

// logit = dot(d2[t], Wd3) + bd3 ; out = sigmoid(logit)
__global__ void k_final(const _Float16* __restrict__ d2, const float* __restrict__ Wd3,
                        const float* __restrict__ bd3, float* __restrict__ out) {
  int t = blockIdx.x * blockDim.x + threadIdx.x;
  if (t < NQ) {
    const _Float16* r = d2 + (long)t * EMB;
    float a = 0.0f;
#pragma unroll
    for (int k = 0; k < EMB; ++k) a = fmaf((float)r[k], Wd3[k], a);
    a += bd3[0];
    out[t] = 1.0f / (1.0f + expf(-a));
  }
}

// ---------------------------------------------------------------------------
// WMMA GEMM:  out[M,NC] = act( A[M,K](f16) @ Bt[NC,K]^T(f16) + bias )
// blockDim=64 (2 waves). Each wave owns one 16-row block: A fragments for the
// full K dimension stay in VGPRs while the wave sweeps all N-tiles, so A is
// read exactly once. B (weights) is tiny and L2/L0 resident.
// M must be a multiple of 32 (exact grids keep EXEC all-ones for WMMA).
// ---------------------------------------------------------------------------
template <int KDIM, int NC, bool RELU, bool OUT_HALF, bool HAS_BIAS>
__global__ __launch_bounds__(64)
void wmma_gemm_kernel(const _Float16* __restrict__ A, const _Float16* __restrict__ Bt,
                      const float* __restrict__ bias, float* __restrict__ outF,
                      _Float16* __restrict__ outH) {
  const int lane = threadIdx.x & 31;
  const long m0 = ((long)blockIdx.x * 2 + (threadIdx.x >> 5)) << 4;

  const _Float16* aPtr = A + (m0 + (lane & 15)) * KDIM;
  v16h afr[KDIM / 32];
#pragma unroll
  for (int kc = 0; kc < KDIM / 32; ++kc) afr[kc] = load_frag(aPtr, kc * 32, lane);

  const long mBase = m0 + ((lane >> 4) << 3);
#pragma unroll
  for (int nt = 0; nt < NC / 16; ++nt) {
    const int n0 = nt << 4;
    const _Float16* bPtr = Bt + (long)(n0 + (lane & 15)) * KDIM;
    v8f acc = {};
#pragma unroll
    for (int kc = 0; kc < KDIM / 32; ++kc) {
      v16h b = load_frag(bPtr, kc * 32, lane);
      acc = __builtin_amdgcn_wmma_f32_16x16x32_f16(false, afr[kc], false, b, (short)0,
                                                   acc, false, false);
    }
    const int col = n0 + (lane & 15);
    const float bv = HAS_BIAS ? bias[col] : 0.0f;
#pragma unroll
    for (int r = 0; r < 8; ++r) {
      float v = acc[r] + bv;
      if (RELU) v = fmaxf(v, 0.0f);
      long idx = (mBase + r) * NC + col;
      if (OUT_HALF) outH[idx] = (_Float16)v;
      else          outF[idx] = v;
    }
  }
}

// Decoder layer 1 with fused pair-gather:
// A row r = concat(z[q0[r]], z[q1[r]]) (128 halves). Each 16B fragment segment
// lies entirely inside one endpoint's 64-half embedding for every k-chunk.
// Gather is done ONCE per wave; fragments reused across all 8 N-tiles.
__global__ __launch_bounds__(64)
void wmma_decoder1_kernel(const _Float16* __restrict__ zh, const int* __restrict__ q,
                          const _Float16* __restrict__ Bt, const float* __restrict__ bias,
                          _Float16* __restrict__ outH) {
  const int lane = threadIdx.x & 31;
  const long m0 = ((long)blockIdx.x * 2 + (threadIdx.x >> 5)) << 4;
  const long r = m0 + (lane & 15);

  const _Float16* za = zh + (long)q[r] * EMB;
  const _Float16* zb = zh + (long)q[NQ + r] * EMB;
  const int off = (lane >> 4) << 3;

  v16h afr[HID / 32];
#pragma unroll
  for (int kc = 0; kc < HID / 32; ++kc) {
    const int k1 = kc * 32 + off;
    const int k2 = kc * 32 + off + 16;
    Frag16 ua;
    ua.f4[0] = *reinterpret_cast<const float4*>((k1 < EMB) ? (za + k1) : (zb + (k1 - EMB)));
    ua.f4[1] = *reinterpret_cast<const float4*>((k2 < EMB) ? (za + k2) : (zb + (k2 - EMB)));
    afr[kc] = ua.h;
  }

  const long mBase = m0 + ((lane >> 4) << 3);
#pragma unroll
  for (int nt = 0; nt < HID / 16; ++nt) {
    const int n0 = nt << 4;
    const _Float16* bPtr = Bt + (long)(n0 + (lane & 15)) * HID;
    v8f acc = {};
#pragma unroll
    for (int kc = 0; kc < HID / 32; ++kc) {
      v16h b = load_frag(bPtr, kc * 32, lane);
      acc = __builtin_amdgcn_wmma_f32_16x16x32_f16(false, afr[kc], false, b, (short)0,
                                                   acc, false, false);
    }
    const int col = n0 + (lane & 15);
    const float bv = bias[col];
#pragma unroll
    for (int rr = 0; rr < 8; ++rr) {
      float v = fmaxf(acc[rr] + bv, 0.0f);
      outH[(mBase + rr) * HID + col] = (_Float16)v;
    }
  }
}

// ---------------------------------------------------------------------------
extern "C" void kernel_launch(void* const* d_in, const int* in_sizes, int n_in,
                              void* d_out, int out_size, void* d_ws, size_t ws_size,
                              hipStream_t stream) {
  const float* x   = (const float*)d_in[0];
  const int*   ei  = (const int*)d_in[1];    // [2, NE] int32
  const int*   qe  = (const int*)d_in[2];    // [2, NQ] int32
  const float* W1  = (const float*)d_in[3];
  const float* b1  = (const float*)d_in[4];
  const float* W2  = (const float*)d_in[5];
  const float* b2  = (const float*)d_in[6];
  const float* Wfc = (const float*)d_in[7];
  const float* bfc = (const float*)d_in[8];
  const float* Wd1 = (const float*)d_in[9];
  const float* bd1 = (const float*)d_in[10];
  const float* Wd2 = (const float*)d_in[11];
  const float* bd2 = (const float*)d_in[12];
  const float* Wd3 = (const float*)d_in[13];
  const float* bd3 = (const float*)d_in[14];
  float* out = (float*)d_out;

  // workspace sub-allocation (256B aligned)
  char* ws = (char*)d_ws;
  size_t o = 0;
  auto take = [&](size_t bytes) -> char* {
    char* p = ws + o;
    o += (bytes + 255) & ~(size_t)255;
    return p;
  };
  float*     dinv = (float*)take((size_t)NN * 4);
  float*     fbuf = (float*)take((size_t)NN * HID * 4);  // h1 then h2
  float*     abuf = (float*)take((size_t)NN * HID * 4);  // agg1 then agg2
  _Float16*  hh   = (_Float16*)take((size_t)NN * HID * 2);
  _Float16*  zh   = (_Float16*)take((size_t)NN * EMB * 2);
  _Float16*  d1h  = (_Float16*)take((size_t)NQ * HID * 2);
  _Float16*  d2h  = (_Float16*)take((size_t)NQ * EMB * 2);
  _Float16*  w2t  = (_Float16*)take((size_t)HID * HID * 2);
  _Float16*  wfct = (_Float16*)take((size_t)EMB * HID * 2);
  _Float16*  wd1t = (_Float16*)take((size_t)HID * HID * 2);
  _Float16*  wd2t = (_Float16*)take((size_t)EMB * HID * 2);

  const int B = 256;
  const long NH = (long)NN * HID;

  // degrees -> dinv
  k_fill1<<<(NN + B - 1) / B, B, 0, stream>>>(dinv, NN);
  k_deg_accum<<<(NE + B - 1) / B, B, 0, stream>>>(ei, dinv);
  k_rsqrt<<<(NN + B - 1) / B, B, 0, stream>>>(dinv, NN);

  // weight conversions (transposed f16)
  k_cvtT<<<(HID * HID + B - 1) / B, B, 0, stream>>>(W2, w2t, HID, HID);
  k_cvtT<<<(HID * EMB + B - 1) / B, B, 0, stream>>>(Wfc, wfct, HID, EMB);
  k_cvtT<<<(HID * HID + B - 1) / B, B, 0, stream>>>(Wd1, wd1t, HID, HID);
  k_cvtT<<<(HID * EMB + B - 1) / B, B, 0, stream>>>(Wd2, wd2t, HID, EMB);

  // --- GCN layer 1 ---
  k_in_gemm<<<(int)(NH / B), B, 0, stream>>>(x, W1, fbuf);
  k_self_init<<<(int)(NH / B), B, 0, stream>>>(fbuf, dinv, b1, abuf);
  k_scatter<<<(int)((long)NE * 32 / B), B, 0, stream>>>(fbuf, dinv, ei, abuf);
  k_relu_cvt<<<(int)(NH / B), B, 0, stream>>>(abuf, hh, NH);

  // --- GCN layer 2: h2 = h1 @ W2 (WMMA), then norm/scatter/relu ---
  wmma_gemm_kernel<HID, HID, false, false, false>
      <<<NN / 32, 64, 0, stream>>>(hh, w2t, nullptr, fbuf, nullptr);
  k_self_init<<<(int)(NH / B), B, 0, stream>>>(fbuf, dinv, b2, abuf);
  k_scatter<<<(int)((long)NE * 32 / B), B, 0, stream>>>(fbuf, dinv, ei, abuf);
  k_relu_cvt<<<(int)(NH / B), B, 0, stream>>>(abuf, hh, NH);

  // --- fc: z = h @ Wfc + bfc  (f16 out for decoder gathers) ---
  wmma_gemm_kernel<HID, EMB, false, true, true>
      <<<NN / 32, 64, 0, stream>>>(hh, wfct, bfc, nullptr, zh);

  // --- decoder ---
  wmma_decoder1_kernel<<<NQ / 32, 64, 0, stream>>>(zh, qe, wd1t, bd1, d1h);
  wmma_gemm_kernel<HID, EMB, true, true, true>
      <<<NQ / 32, 64, 0, stream>>>(d1h, wd2t, bd2, nullptr, d2h);
  k_final<<<(NQ + B - 1) / B, B, 0, stream>>>(d2h, Wd3, bd3, out);
}
